// MoEFFN_83811991814246
// MI455X (gfx1250) — matrix operand order repaired and checked by reading, not compiled, not run on hardware
//
#include <hip/hip_runtime.h>
#include <hip/hip_bf16.h>

// ---- Problem constants (match reference) ----
#define D_MODEL 1024
#define D_INNER 4096
#define NEXP    8
#define NTOK    8192        // B*S = 4*2048
#define LDSTR   40          // LDS row stride (K=32 + 8 pad): 80B rows, 16B-aligned frags

typedef __attribute__((ext_vector_type(16))) __bf16 v16bf;
typedef __attribute__((ext_vector_type(8)))  float  v8f;
typedef unsigned int u32x4 __attribute__((ext_vector_type(4)));
typedef int          i32x4 __attribute__((ext_vector_type(4)));
typedef int          i32x8 __attribute__((ext_vector_type(8)));

__device__ __forceinline__ __bf16 f2bf(float f) {
    unsigned u = __builtin_bit_cast(unsigned, f);
    unsigned short s = (unsigned short)(u >> 16);   // truncate to bf16
    return __builtin_bit_cast(__bf16, s);
}

// Pack two fp32 -> one bf16 pair (truncation) with a single v_perm_b32.
__device__ __forceinline__ unsigned pack2bf(float f0, float f1) {
    unsigned u0 = __builtin_bit_cast(unsigned, f0);
    unsigned u1 = __builtin_bit_cast(unsigned, f1);
    // dst bytes: [0]=u0.b2 [1]=u0.b3 [2]=u1.b2 [3]=u1.b3  (S1=u0 -> sel 0..3, S0=u1 -> sel 4..7)
    return __builtin_amdgcn_perm(u1, u0, 0x07060302u);
}

// A-matrix 16x32 bf16 fragment (ISA 7.12.2): lane L holds row M=L%16;
// elems 0..7 -> K = g*8 + i ; elems 8..15 -> K = 16 + g*8 + (i-8), g = L/16.
__device__ __forceinline__ v16bf load_frag_a(const __bf16* tile, int row, int g) {
    const __bf16* p = tile + row * LDSTR + g * 8;
    v16bf f;
#pragma unroll
    for (int i = 0; i < 8; ++i) f[i] = p[i];
#pragma unroll
    for (int i = 0; i < 8; ++i) f[8 + i] = p[16 + i];
    return f;
}

// B-matrix 32x16 bf16 fragment: same K order as A (elems 0..7 -> K = g*8 + i ; elems 8..15 -> K = 16 + g*8 + (i-8)).
__device__ __forceinline__ v16bf load_frag_b(const __bf16* tile, int col, int g) {
    const __bf16* p = tile + col * LDSTR + g * 8;
    v16bf f;
#pragma unroll
    for (int i = 0; i < 8; ++i) f[i] = p[i];
#pragma unroll
    for (int i = 0; i < 8; ++i) f[8 + i] = p[16 + i];
    return f;
}

// ---------------- Phase A: router (logits -> softmax top-2 -> expert lists) ----------------
__global__ void router_topk(const float* __restrict__ x, const float* __restrict__ rw,
                            const float* __restrict__ rb, int* __restrict__ counts,
                            int* __restrict__ tlist, float* __restrict__ wlist) {
    int t = blockIdx.x * blockDim.x + threadIdx.x;
    if (t >= NTOK) return;
    float lg[NEXP];
#pragma unroll
    for (int e = 0; e < NEXP; ++e) lg[e] = rb[e];
    const float* xr = x + (size_t)t * D_MODEL;
    for (int d = 0; d < D_MODEL; d += 4) {
        float x0 = xr[d], x1 = xr[d + 1], x2 = xr[d + 2], x3 = xr[d + 3];
#pragma unroll
        for (int e = 0; e < NEXP; ++e) {
            const float* w = rw + e * D_MODEL + d;
            lg[e] += x0 * w[0] + x1 * w[1] + x2 * w[2] + x3 * w[3];
        }
    }
    int i0 = 0;
#pragma unroll
    for (int e = 1; e < NEXP; ++e) if (lg[e] > lg[i0]) i0 = e;
    int i1 = (i0 == 0) ? 1 : 0;
#pragma unroll
    for (int e = 0; e < NEXP; ++e) if (e != i0 && lg[e] > lg[i1]) i1 = e;
    float p1 = __expf(lg[i1] - lg[i0]);          // p0 = 1
    float inv = 1.0f / (1.0f + p1);
    int pos0 = atomicAdd(&counts[i0], 1);
    tlist[i0 * NTOK + pos0] = t;  wlist[i0 * NTOK + pos0] = inv;
    int pos1 = atomicAdd(&counts[i1], 1);
    tlist[i1 * NTOK + pos1] = t;  wlist[i1 * NTOK + pos1] = p1 * inv;
}

// ---------------- exclusive scan over 8 expert counts (compact h slots) ----------------
__global__ void exscan8(const int* __restrict__ counts, int* __restrict__ offs) {
    if (threadIdx.x == 0) {
        int s = 0;
#pragma unroll
        for (int e = 0; e < NEXP; ++e) { offs[e] = s; s += counts[e]; }
    }
}

// ---------------- Phase B: gate & up GEMMs + SiLU, h -> bf16 workspace ----------------
// grid: (D_INNER/128, NTOK/64, NEXP), block 256 (8 waves: 2 M x 4 N)
__global__ __launch_bounds__(256, 2)
void gateup_kernel(const float* __restrict__ x, const float* __restrict__ gw,
                   const float* __restrict__ uw, const int* __restrict__ counts,
                   const int* __restrict__ offs, const int* __restrict__ tlist,
                   __bf16* __restrict__ hbuf) {
    const int e = blockIdx.z;
    const int cnt = counts[e];
    const int m0 = blockIdx.y * 64;
    if (m0 >= cnt) return;
    const int n0 = blockIdx.x * 128;

    __shared__ __bf16 As[64 * LDSTR];
    __shared__ __bf16 Bg[128 * LDSTR];
    __shared__ __bf16 Bu[128 * LDSTR];
    __shared__ int toks[64];

    const int tid = threadIdx.x;
    if (tid < 64) {
        int m = m0 + tid;
        toks[tid] = (m < cnt) ? tlist[e * NTOK + m] : -1;
    }
    __syncthreads();

    const int wave = tid >> 5, lane = tid & 31;
    const int g = lane >> 4, ln = lane & 15;
    const int wm = wave >> 2, wn = wave & 3;

    v8f accg[2][2] = {};
    v8f accu[2][2] = {};

    const int arow = tid >> 2, akb = (tid & 3) * 8;     // A: 64 rows x 32 k, 8 elems/thread
    const int brow = tid >> 1, bkb = (tid & 1) * 16;    // B: 128 rows x 32 k, 16 elems/thread
    const size_t wbase = (size_t)e * D_INNER * D_MODEL;
    const int atok = toks[arow];
    const float* ga = gw + wbase + (size_t)(n0 + brow) * D_MODEL + bkb;
    const float* ua = uw + wbase + (size_t)(n0 + brow) * D_MODEL + bkb;
    unsigned* Adst = (unsigned*)&As[arow * LDSTR + akb];
    unsigned* Bgdst = (unsigned*)&Bg[brow * LDSTR + bkb];
    unsigned* Budst = (unsigned*)&Bu[brow * LDSTR + bkb];

    for (int kk = 0; kk < D_MODEL; kk += 32) {
        // stage A (gathered token rows, fp32 -> bf16 via v_perm packing)
        if (atok >= 0) {
            const float* p = x + (size_t)atok * D_MODEL + kk + akb;
#pragma unroll
            for (int i = 0; i < 4; ++i) Adst[i] = pack2bf(p[2 * i], p[2 * i + 1]);
        } else {
#pragma unroll
            for (int i = 0; i < 4; ++i) Adst[i] = 0u;
        }
        // stage gate/up weight tiles
        {
            const float* pg = ga + kk;
            const float* pu = ua + kk;
            __builtin_prefetch(pg + 32, 0, 0);
            __builtin_prefetch(pu + 32, 0, 0);
#pragma unroll
            for (int i = 0; i < 8; ++i) Bgdst[i] = pack2bf(pg[2 * i], pg[2 * i + 1]);
#pragma unroll
            for (int i = 0; i < 8; ++i) Budst[i] = pack2bf(pu[2 * i], pu[2 * i + 1]);
        }
        __syncthreads();

        v16bf a[2], bgf[2], buf[2];
#pragma unroll
        for (int sm = 0; sm < 2; ++sm) a[sm] = load_frag_a(As, wm * 32 + sm * 16 + ln, g);
#pragma unroll
        for (int sn = 0; sn < 2; ++sn) {
            bgf[sn] = load_frag_b(Bg, wn * 32 + sn * 16 + ln, g);
            buf[sn] = load_frag_b(Bu, wn * 32 + sn * 16 + ln, g);
        }
#pragma unroll
        for (int sm = 0; sm < 2; ++sm)
#pragma unroll
            for (int sn = 0; sn < 2; ++sn) {
                accg[sm][sn] = __builtin_amdgcn_wmma_f32_16x16x32_bf16(
                    false, a[sm], false, bgf[sn], (short)0, accg[sm][sn], false, false);
                accu[sm][sn] = __builtin_amdgcn_wmma_f32_16x16x32_bf16(
                    false, a[sm], false, buf[sn], (short)0, accu[sm][sn], false, false);
            }
        __syncthreads();
    }

    // epilogue: h = silu(g) * u -> bf16 workspace at compact slot (offs[e] + list pos)
    const int base = offs[e];
#pragma unroll
    for (int sm = 0; sm < 2; ++sm)
#pragma unroll
        for (int sn = 0; sn < 2; ++sn)
#pragma unroll
            for (int r = 0; r < 8; ++r) {
                int Ml = wm * 32 + sm * 16 + r + 8 * g;
                if (m0 + Ml < cnt) {
                    int Nl = wn * 32 + sn * 16 + ln;
                    float gv = accg[sm][sn][r], uv = accu[sm][sn][r];
                    float h = (gv / (1.0f + __expf(-gv))) * uv;
                    hbuf[(size_t)(base + m0 + Ml) * D_INNER + (n0 + Nl)] = f2bf(h);
                }
            }
}

// ---------------- Phase C: down GEMM + weighted atomic combine into out ----------------
// grid: (D_MODEL/128, NTOK/64, NEXP), block 256
__global__ __launch_bounds__(256, 2)
void down_kernel(const __bf16* __restrict__ hbuf, const float* __restrict__ dw,
                 const int* __restrict__ counts, const int* __restrict__ offs,
                 const int* __restrict__ tlist, const float* __restrict__ wlist,
                 float* __restrict__ out) {
    const int e = blockIdx.z;
    const int cnt = counts[e];
    const int m0 = blockIdx.y * 64;
    if (m0 >= cnt) return;
    const int n0 = blockIdx.x * 128;
    const int base = offs[e];

    __shared__ __bf16 As[64 * LDSTR];
    __shared__ __bf16 Bd[128 * LDSTR];

    const int tid = threadIdx.x;
    const int wave = tid >> 5, lane = tid & 31;
    const int g = lane >> 4, ln = lane & 15;
    const int wm = wave >> 2, wn = wave & 3;

    v8f acc[2][2] = {};

    const int brow = tid >> 1, bkb = (tid & 1) * 16;
    const float* da = dw + (size_t)e * D_MODEL * D_INNER + (size_t)(n0 + brow) * D_INNER + bkb;
    unsigned* Bdst = (unsigned*)&Bd[brow * LDSTR + bkb];

#if __has_builtin(__builtin_amdgcn_tensor_load_to_lds)
    // ---- TDM path: DMA the (already-bf16) h tile 64x32 into padded LDS rows ----
    // D# group0: count=1 | lds_addr | 57-bit global addr | type=2
    // D# group1: data_size=2B, pad after 16 DWORDs by 4 DWORDs (80B LDS rows),
    //            tensor_dim0=4096, tensor_dim1=cnt-m0 (OOB rows read as zero),
    //            tile 32x64, tensor_dim0_stride=4096.
    const unsigned lds_off = (unsigned)(size_t)(void*)&As[0];
    const unsigned rows_avail = (unsigned)(cnt - m0);
    const unsigned long long gbase =
        (unsigned long long)(size_t)(hbuf + (size_t)(base + m0) * D_INNER);
    i32x8 g1;
    g1[0] = (int)((1u << 16) | (1u << 20) | (3u << 22) | (3u << 25));
    g1[1] = (int)((unsigned)(D_INNER & 0xFFFF) << 16);
    g1[2] = (int)((rows_avail & 0xFFFFu) << 16);
    g1[3] = (int)((rows_avail >> 16) | (32u << 16));
    g1[4] = 64;
    g1[5] = D_INNER;
    g1[6] = 0;
    g1[7] = 0;
    i32x4 gz = {0, 0, 0, 0};
#else
    const int arow = tid >> 2, akb = (tid & 3) * 8;
    const bool arow_ok = (m0 + arow) < cnt;
    const __bf16* ha = hbuf + (size_t)(base + m0 + arow) * D_INNER + akb;
    unsigned* Adst = (unsigned*)&As[arow * LDSTR + akb];
#endif

    for (int kk = 0; kk < D_INNER; kk += 32) {
#if __has_builtin(__builtin_amdgcn_tensor_load_to_lds)
        if (wave == 0) {
            unsigned long long ga64 = gbase + (unsigned)(kk * 2);
            u32x4 g0;
            g0[0] = 1u;                                   // count=1, user mode
            g0[1] = lds_off;
            g0[2] = (unsigned)(ga64 & 0xFFFFFFFFull);
            g0[3] = (unsigned)((ga64 >> 32) & 0x01FFFFFFull) | (2u << 30);  // type=2
#if __clang_major__ >= 23
            __builtin_amdgcn_tensor_load_to_lds(g0, g1, gz, gz, g1, 0);
#else
            __builtin_amdgcn_tensor_load_to_lds(g0, g1, gz, gz, 0);
#endif
            __builtin_amdgcn_s_wait_tensorcnt(0);
        }
#else
        if (arow_ok) {
            const unsigned* p = (const unsigned*)(ha + kk);
#pragma unroll
            for (int i = 0; i < 4; ++i) Adst[i] = p[i];
        } else {
#pragma unroll
            for (int i = 0; i < 4; ++i) Adst[i] = 0u;
        }
#endif
        {
            const float* pd = da + kk;
            __builtin_prefetch(pd + 32, 0, 0);
#pragma unroll
            for (int i = 0; i < 8; ++i) Bdst[i] = pack2bf(pd[2 * i], pd[2 * i + 1]);
        }
        __syncthreads();

        v16bf a[2], b[2];
#pragma unroll
        for (int sm = 0; sm < 2; ++sm) a[sm] = load_frag_a(As, wm * 32 + sm * 16 + ln, g);
#pragma unroll
        for (int sn = 0; sn < 2; ++sn) b[sn] = load_frag_b(Bd, wn * 32 + sn * 16 + ln, g);
#pragma unroll
        for (int sm = 0; sm < 2; ++sm)
#pragma unroll
            for (int sn = 0; sn < 2; ++sn)
                acc[sm][sn] = __builtin_amdgcn_wmma_f32_16x16x32_bf16(
                    false, a[sm], false, b[sn], (short)0, acc[sm][sn], false, false);
        __syncthreads();
    }

    // epilogue: out[token] += w * y  (each token touched by exactly its 2 experts)
#pragma unroll
    for (int sm = 0; sm < 2; ++sm)
#pragma unroll
        for (int sn = 0; sn < 2; ++sn)
#pragma unroll
            for (int r = 0; r < 8; ++r) {
                int Ml = wm * 32 + sm * 16 + r + 8 * g;
                if (m0 + Ml < cnt) {
                    int pos = e * NTOK + m0 + Ml;
                    int tok = tlist[pos];
                    float w = wlist[pos];
                    int Nl = wn * 32 + sn * 16 + ln;
                    atomicAdd(&out[(size_t)tok * D_MODEL + (n0 + Nl)], w * acc[sm][sn][r]);
                }
            }
}

// ---------------- launch ----------------
extern "C" void kernel_launch(void* const* d_in, const int* in_sizes, int n_in,
                              void* d_out, int out_size, void* d_ws, size_t ws_size,
                              hipStream_t stream) {
    const float* x  = (const float*)d_in[0];
    const float* rw = (const float*)d_in[1];
    const float* rb = (const float*)d_in[2];
    const float* gw = (const float*)d_in[3];
    const float* uw = (const float*)d_in[4];
    const float* dw = (const float*)d_in[5];
    float* out = (float*)d_out;

    // workspace layout
    char* ws = (char*)d_ws;
    int*   counts = (int*)ws;                         // 8 ints
    int*   offs   = (int*)(ws + 32);                  // 8 ints
    int*   tlist  = (int*)(ws + 64);                  // NEXP*NTOK ints
    float* wlist  = (float*)(ws + 64 + (size_t)NEXP * NTOK * 4);
    size_t hoff   = 64 + (size_t)NEXP * NTOK * 8;
    hoff = (hoff + 255) & ~(size_t)255;
    __bf16* hbuf  = (__bf16*)(ws + hoff);             // up to NTOK*2 slots x D_INNER bf16

    hipMemsetAsync(counts, 0, 32, stream);
    hipMemsetAsync(out, 0, (size_t)out_size * sizeof(float), stream);

    router_topk<<<NTOK / 256, 256, 0, stream>>>(x, rw, rb, counts, tlist, wlist);
    exscan8<<<1, 32, 0, stream>>>(counts, offs);
    gateup_kernel<<<dim3(D_INNER / 128, NTOK / 64, NEXP), 256, 0, stream>>>(
        x, gw, uw, counts, offs, tlist, hbuf);
    down_kernel<<<dim3(D_MODEL / 128, NTOK / 64, NEXP), 256, 0, stream>>>(
        hbuf, dw, counts, offs, tlist, wlist, out);
}
